// BotRGCN_39487929319593
// MI455X (gfx1250) — matrix operand order repaired and checked by reading, not compiled, not run on hardware
//
#include <hip/hip_runtime.h>
#include <hip/hip_bf16.h>

typedef __attribute__((ext_vector_type(16))) __bf16 v16bf;
typedef __attribute__((ext_vector_type(8)))  float  v8f;
typedef __attribute__((ext_vector_type(8)))  unsigned short v8us;
typedef __attribute__((ext_vector_type(4)))  unsigned short v4us;

#define HIDDEN 128
#define LDS_STRIDE 136   // 128 + 8 pad: bank-conflict-free B-fragment loads
#define SEG_SZ (128 * LDS_STRIDE)

// f32 -> bf16 via native type (backend picks v_cvt_pk_bf16_f32 if available)
__device__ __forceinline__ unsigned short bf16u(float f) {
    union { __bf16 b; unsigned short s; } t;
    t.b = (__bf16)f;
    return t.s;
}
__device__ __forceinline__ unsigned int pack2(float lo, float hi) {
    union { __bf16 b[2]; unsigned int u; } t;
    t.b[0] = (__bf16)lo;
    t.b[1] = (__bf16)hi;
    return t.u;
}
__device__ __forceinline__ float bf2f(unsigned short s) {
    return __uint_as_float((unsigned int)s << 16);   // exact widening
}

// ---------------------------------------------------------------------------
// Per-node feature embed: E[n, 0:64)  = leaky(x_num @ W_num + b_num)
//                         E[n,64:128) = leaky(x_cat @ W_cat + b_cat)
// Output stored directly as bf16 (GEMM input format).
// ---------------------------------------------------------------------------
__global__ __launch_bounds__(256) void embed_kernel(
    const float* __restrict__ x, const float* __restrict__ Wn,
    const float* __restrict__ bn, const float* __restrict__ Wc,
    const float* __restrict__ bc, unsigned short* __restrict__ E, int M)
{
    int t = blockIdx.x * 256 + threadIdx.x;
    int node = t >> 7;
    int j = t & 127;
    if (node >= M) return;
    const float* xr = x + (size_t)node * 8;
    float acc;
    if (j < 64) {
        acc = bn[j];
        #pragma unroll
        for (int k = 0; k < 5; ++k) acc += xr[k] * Wn[k * 64 + j];
    } else {
        int jj = j - 64;
        acc = bc[jj];
        #pragma unroll
        for (int k = 0; k < 3; ++k) acc += xr[5 + k] * Wc[k * 64 + jj];
    }
    acc = acc >= 0.0f ? acc : 0.01f * acc;   // leaky_relu slope 0.01
    E[(size_t)node * HIDDEN + j] = bf16u(acc);
}

// ---------------------------------------------------------------------------
// Edge scatter: one wave32 per edge. h is bf16 (256 B/row random gather,
// L2-resident); accumulation stays f32 via hardware global atomics.
//   agg[type][dst][:] += h[src][:]   ;   deg[type][dst] += 1
// ---------------------------------------------------------------------------
__global__ __launch_bounds__(256) void edge_scatter_kernel(
    const unsigned short* __restrict__ h, const int* __restrict__ ei,
    const int* __restrict__ et, float* __restrict__ agg,
    float* __restrict__ deg, int nE, int M)
{
    int wid  = (blockIdx.x * 256 + threadIdx.x) >> 5;
    int lane = threadIdx.x & 31;
    if (wid >= nE) return;
    int src = ei[wid];
    int dst = ei[nE + wid];
    int r   = et[wid];
    const v4us v = *(const v4us*)(h + (size_t)src * HIDDEN + lane * 4);
    float* base = agg + (((size_t)r * M + dst) * HIDDEN) + lane * 4;
    unsafeAtomicAdd(base + 0, bf2f(v[0]));
    unsafeAtomicAdd(base + 1, bf2f(v[1]));
    unsafeAtomicAdd(base + 2, bf2f(v[2]));
    unsafeAtomicAdd(base + 3, bf2f(v[3]));
    if (lane == 0) unsafeAtomicAdd(deg + (size_t)r * M + dst, 1.0f);
}

// ---------------------------------------------------------------------------
// Fused node GEMM:
//   out = A0@W0 [+ (A1/max(deg0,1))@W1 + (A2/max(deg1,1))@W2] + bias [+prelu]
// A0 is bf16 (raw fragment loads); A1/A2 are f32 agg buffers (converted +
// per-row 1/deg scaled at load). out is f32 (outf) or bf16 (outb).
// Block = 256 threads = 8 waves; tile = 128 rows x 128 cols.
// Weights staged once to LDS transposed (bwT[n][k], bf16, stride 136); all 8
// B fragments of a K-chunk are fetched before the 8 WMMAs (pipelined dscnt).
// ---------------------------------------------------------------------------
__global__ __launch_bounds__(256) void wmma_gemm_kernel(
    const unsigned short* __restrict__ A0,
    const float* __restrict__ A1, const float* __restrict__ A2,
    const float* __restrict__ deg,
    const float* __restrict__ W0, const float* __restrict__ W1,
    const float* __restrict__ W2, const float* __restrict__ bias,
    const float* __restrict__ prelu_a,
    float* __restrict__ outf, unsigned short* __restrict__ outb,
    int M, int nseg)
{
    extern __shared__ unsigned short bwT[];   // [nseg][128 n][136 k] bf16

    // ---- stage weights: W[k][n] f32 -> bwT[n*136+k] bf16 -------------------
    for (int idx = threadIdx.x; idx < nseg * 128 * 128; idx += 256) {
        int seg = idx >> 14;          // idx / 16384
        int rem = idx & 16383;
        int k = rem & 127;            // k fastest: contiguous LDS writes
        int n = rem >> 7;
        const float* W = (seg == 0) ? W0 : ((seg == 1) ? W1 : W2);
        bwT[seg * SEG_SZ + n * LDS_STRIDE + k] = bf16u(W[k * 128 + n]);
    }
    __syncthreads();

    const int wave = threadIdx.x >> 5;
    const int lane = threadIdx.x & 31;
    const int hi   = (lane & 16) ? 1 : 0;            // lane half
    const int mtile = blockIdx.x * 128 + wave * 16;
    const int rowm  = mtile + (lane & 15);           // this lane's A row
    const int rowc  = rowm < M ? rowm : (M - 1);     // clamped: branch-free loads

    v8f acc[8] = {};

    // ======== segment 0: bf16 A, scale == 1 (raw fragment loads) ===========
    {
        const unsigned short* arow = A0 + (size_t)rowc * HIDDEN;
        #pragma unroll
        for (int kc = 0; kc < 4; ++kc) {
            // A fragment (16x32 bf16): lanes 0-15 K={0..7,16..23}, lanes 16-31 +8
            int kA = kc * 32 + (hi ? 8 : 0);
            union { v8us p[2]; v16bf v; } af;
            af.p[0] = *(const v8us*)(arow + kA);        // K kA..kA+7
            af.p[1] = *(const v8us*)(arow + kA + 16);   // K kA+16..kA+23

            int k0 = kc * 32 + (hi ? 8 : 0);
            union { v8us p[2]; v16bf v; } bf[8];
            #pragma unroll
            for (int nt = 0; nt < 8; ++nt) {
                const unsigned short* bp = bwT + (nt * 16 + (lane & 15)) * LDS_STRIDE + k0;
                bf[nt].p[0] = *(const v8us*)(bp);
                bf[nt].p[1] = *(const v8us*)(bp + 16);
            }
            #pragma unroll
            for (int nt = 0; nt < 8; ++nt) {
                acc[nt] = __builtin_amdgcn_wmma_f32_16x16x32_bf16(
                    false, af.v, false, bf[nt].v, (short)0, acc[nt], false, false);
            }
        }
    }

    // ======== segments 1..nseg-1: f32 agg, per-row 1/deg scaling ============
    for (int seg = 1; seg < nseg; ++seg) {
        const float* A = (seg == 1) ? A1 : A2;
        float d = deg[(size_t)(seg - 1) * M + rowc];
        float scale = 1.0f / fmaxf(d, 1.0f);
        const unsigned short* bw = bwT + seg * SEG_SZ;
        const float* arow = A + (size_t)rowc * HIDDEN;

        #pragma unroll
        for (int kc = 0; kc < 4; ++kc) {
            int kA = kc * 32 + (hi ? 8 : 0);
            float4 a0 = *(const float4*)(arow + kA + 0);
            float4 a1 = *(const float4*)(arow + kA + 4);
            float4 a2 = *(const float4*)(arow + kA + 16);
            float4 a3 = *(const float4*)(arow + kA + 20);
            union { unsigned int u[8]; v16bf v; } af;
            af.u[0] = pack2(a0.x * scale, a0.y * scale);
            af.u[1] = pack2(a0.z * scale, a0.w * scale);
            af.u[2] = pack2(a1.x * scale, a1.y * scale);
            af.u[3] = pack2(a1.z * scale, a1.w * scale);
            af.u[4] = pack2(a2.x * scale, a2.y * scale);
            af.u[5] = pack2(a2.z * scale, a2.w * scale);
            af.u[6] = pack2(a3.x * scale, a3.y * scale);
            af.u[7] = pack2(a3.z * scale, a3.w * scale);

            int k0 = kc * 32 + (hi ? 8 : 0);
            union { v8us p[2]; v16bf v; } bf[8];
            #pragma unroll
            for (int nt = 0; nt < 8; ++nt) {
                const unsigned short* bp = bw + (nt * 16 + (lane & 15)) * LDS_STRIDE + k0;
                bf[nt].p[0] = *(const v8us*)(bp);
                bf[nt].p[1] = *(const v8us*)(bp + 16);
            }
            #pragma unroll
            for (int nt = 0; nt < 8; ++nt) {
                acc[nt] = __builtin_amdgcn_wmma_f32_16x16x32_bf16(
                    false, af.v, false, bf[nt].v, (short)0, acc[nt], false, false);
            }
        }
    }

    // ---- epilogue: bias (+ optional prelu), guarded scatter to out ---------
    #pragma unroll
    for (int nt = 0; nt < 8; ++nt) {
        int n = nt * 16 + (lane & 15);
        float b = bias[n];
        float a = prelu_a ? prelu_a[n] : 0.0f;
        #pragma unroll
        for (int r = 0; r < 8; ++r) {
            int row = mtile + r + (hi ? 8 : 0);
            if (row < M) {
                float v = acc[nt][r] + b;
                if (prelu_a) v = (v >= 0.0f) ? v : a * v;
                if (outb) outb[(size_t)row * HIDDEN + n] = bf16u(v);
                else      outf[(size_t)row * HIDDEN + n] = v;
            }
        }
    }
}

// ---------------------------------------------------------------------------
extern "C" void kernel_launch(void* const* d_in, const int* in_sizes, int n_in,
                              void* d_out, int out_size, void* d_ws, size_t ws_size,
                              hipStream_t stream)
{
    const float* x       = (const float*)d_in[0];
    const int*   ei      = (const int*)  d_in[1];
    const int*   et      = (const int*)  d_in[2];
    const float* W_num   = (const float*)d_in[3];
    const float* b_num   = (const float*)d_in[4];
    const float* W_cat   = (const float*)d_in[5];
    const float* b_cat   = (const float*)d_in[6];
    const float* W_in    = (const float*)d_in[7];
    const float* b_in    = (const float*)d_in[8];
    const float* prelu_a = (const float*)d_in[9];
    const float* W_rel1  = (const float*)d_in[10];
    const float* W_root1 = (const float*)d_in[11];
    const float* b_rgcn1 = (const float*)d_in[12];
    const float* W_rel2  = (const float*)d_in[13];
    const float* W_root2 = (const float*)d_in[14];
    const float* b_rgcn2 = (const float*)d_in[15];
    const float* W_cls   = (const float*)d_in[16];
    const float* b_cls   = (const float*)d_in[17];

    const int M  = in_sizes[0] / 8;      // 100000 nodes
    const int nE = in_sizes[2];          // 1.6M edges

    // workspace: [agg f32 2*M*128][deg f32 2*M][hA bf16 M*128][hB bf16 M*128]
    float* agg = (float*)d_ws;
    float* deg = agg + (size_t)2 * M * HIDDEN;
    unsigned short* hA = (unsigned short*)(deg + (size_t)2 * M);
    unsigned short* hB = hA + (size_t)M * HIDDEN;
    const size_t aggdeg_bytes = ((size_t)2 * M * HIDDEN + (size_t)2 * M) * sizeof(float);

    const size_t lds1 = (size_t)1 * SEG_SZ * sizeof(unsigned short);  // 34.8 KB
    const size_t lds3 = (size_t)3 * SEG_SZ * sizeof(unsigned short);  // 104 KB
    dim3 gemm_grid((M + 127) / 128);
    dim3 edge_grid(((size_t)nE * 32 + 255) / 256);

    // 1) embed: x -> E (hA, bf16)
    embed_kernel<<<((size_t)M * 128 + 255) / 256, 256, 0, stream>>>(
        x, W_num, b_num, W_cat, b_cat, hA, M);

    // 2) H0 = prelu(E @ W_in + b_in) -> hB (bf16)
    wmma_gemm_kernel<<<gemm_grid, 256, lds1, stream>>>(
        hA, nullptr, nullptr, nullptr, W_in, nullptr, nullptr,
        b_in, prelu_a, nullptr, hB, M, 1);

    // 3) layer 1 aggregation (f32 atomics into L2-resident agg)
    hipMemsetAsync(agg, 0, aggdeg_bytes, stream);
    edge_scatter_kernel<<<edge_grid, 256, 0, stream>>>(hB, ei, et, agg, deg, nE, M);

    // 4) H1 = H0@W_root1 + (agg0/deg0)@W_rel1[0] + (agg1/deg1)@W_rel1[1] + b -> hA
    wmma_gemm_kernel<<<gemm_grid, 256, lds3, stream>>>(
        hB, agg, agg + (size_t)M * HIDDEN, deg,
        W_root1, W_rel1, W_rel1 + 128 * 128, b_rgcn1, nullptr, nullptr, hA, M, 3);

    // 5) layer 2 aggregation
    hipMemsetAsync(agg, 0, aggdeg_bytes, stream);
    edge_scatter_kernel<<<edge_grid, 256, 0, stream>>>(hA, ei, et, agg, deg, nE, M);

    // 6) H2 -> hB (bf16)
    wmma_gemm_kernel<<<gemm_grid, 256, lds3, stream>>>(
        hA, agg, agg + (size_t)M * HIDDEN, deg,
        W_root2, W_rel2, W_rel2 + 128 * 128, b_rgcn2, nullptr, nullptr, hB, M, 3);

    // 7) out = H2 @ W_cls + b_cls  (f32 output)
    wmma_gemm_kernel<<<gemm_grid, 256, lds1, stream>>>(
        hB, nullptr, nullptr, nullptr, W_cls, nullptr, nullptr,
        b_cls, nullptr, (float*)d_out, nullptr, M, 1);
}
